// MultiHeadAttention_41283225649701
// MI455X (gfx1250) — matrix operand order repaired and checked by reading, not compiled, not run on hardware
//
#include <hip/hip_runtime.h>
#include <hip/hip_bf16.h>
#include <math.h>

// ---------------------------------------------------------------------------
// Multi-head attention (B=2, L=2048, H=16, Dk=Dv=64, Dm=1024) for gfx1250.
// bf16 WMMA 16x16x32 (f32 accum); wave32; scores computed TRANSPOSED (S^T =
// K.Q^T) so softmax stats reduce in-lane (+1 SWAPX16) instead of 4 shuffle
// rounds; attn emitted via LDS as coalesced non-temporal 128-bit stores.
// ---------------------------------------------------------------------------

#define B_    2
#define L_    2048
#define H_    16
#define DK_   64
#define DV_   64
#define DM_   1024
#define LN_EPS 1e-5f
#define PST   36   // LDS P-buffer row stride (floats): bank-conflict-free, 16B rows

typedef __attribute__((ext_vector_type(16))) __bf16 bf16x16;
typedef __attribute__((ext_vector_type(8)))  float  f32x8;
typedef __attribute__((ext_vector_type(4)))  float  f32x4;

__device__ __forceinline__ f32x8 wmma_bf16(bf16x16 a, bf16x16 b, f32x8 c) {
  // (neg_a, A, neg_b, B, c_mod, C, reuse_a, reuse_b)
  return __builtin_amdgcn_wmma_f32_16x16x32_bf16(false, a, false, b, (short)0, c,
                                                 false, false);
}

// exchange with the other 16-lane half: ds_swizzle SWAPX16 (xor=0x10,and=0x1f)
__device__ __forceinline__ float swap16(float x) {
  return __int_as_float(__builtin_amdgcn_ds_swizzle(__float_as_int(x), 0x401F));
}

// ---------------------------------------------------------------------------
// Kernel 1: attention. One wave = one (b, h, 16-row q stripe).
// ---------------------------------------------------------------------------
__global__ void __launch_bounds__(128)
attn_kernel(const float* __restrict__ q, const float* __restrict__ k,
            const float* __restrict__ v, float* __restrict__ attn,
            float* __restrict__ heads) {
  const int w    = threadIdx.x >> 5;                 // wave in block (0..3)
  const int lane = threadIdx.x & 31;
  const int lo   = lane & 15;
  const int hi   = lane >> 4;

  const int stripe = blockIdx.x * 4 + w;             // 4096 stripes total
  const int qt = stripe & 127;
  const int h  = (stripe >> 7) & 15;
  const int b  = stripe >> 11;
  const int q0 = qt * 16;

  __shared__ float pbuf[4][16 * PST];                // 9 KB
  float* pw = pbuf[w];

  // ---- Q^T B-fragments (two 32-deep d-chunks), 1/sqrt(64) folded in ----
  bf16x16 bq[2];
  {
    const float* qrow = q + ((size_t)b * L_ + q0 + lo) * DM_ + h * DK_;
    #pragma unroll
    for (int c = 0; c < 2; ++c) {
      #pragma unroll
      for (int e = 0; e < 16; ++e)
        bq[c][e] = (__bf16)(qrow[c * 32 + hi * 8 + (e & 7) + ((e >> 3) << 4)] * 0.125f);
    }
  }

  const float* kbase = k + (size_t)b * L_ * DM_ + h * DK_;
  const float* vbase = v + (size_t)b * L_ * DM_ + h * DV_;

  // ---- pass 1: online row max / sum.  Lane owns q-row `lo` (dup per half).
  float mrun = -INFINITY, lrun = 0.0f;
  for (int j = 0; j < L_ / 16; ++j) {
    bf16x16 ka0, ka1;                                // K tile as A-fragments
    {
      const float* kp = kbase + (size_t)(j * 16 + lo) * DM_;
      #pragma unroll
      for (int e = 0; e < 8; ++e) {
        ka0[e]     = (__bf16)kp[hi * 8 + e];
        ka0[8 + e] = (__bf16)kp[16 + hi * 8 + e];
        ka1[e]     = (__bf16)kp[32 + hi * 8 + e];
        ka1[8 + e] = (__bf16)kp[48 + hi * 8 + e];
      }
    }
    f32x8 c = {};
    c = wmma_bf16(ka0, bq[0], c);                    // S^T tile: 16k x 16q
    c = wmma_bf16(ka1, bq[1], c);

    float vmax = c[0];
    #pragma unroll
    for (int r = 1; r < 8; ++r) vmax = fmaxf(vmax, c[r]);
    vmax = fmaxf(vmax, swap16(vmax));
    const float mn = fmaxf(mrun, vmax);
    float ps = 0.0f;
    #pragma unroll
    for (int r = 0; r < 8; ++r) ps += __expf(c[r] - mn);
    ps += swap16(ps);
    lrun = lrun * __expf(mrun - mn) + ps;
    mrun = mn;
  }
  const float rcpl = 1.0f / lrun;

  // ---- pass 2: recompute, emit normalized attn, accumulate O = P @ V ----
  f32x8 oacc[4] = {};
  float* attn_base = attn + (((size_t)(b * H_ + h)) * L_ + q0) * (size_t)L_;

  for (int jc = 0; jc < L_ / 32; ++jc) {
    #pragma unroll
    for (int t = 0; t < 2; ++t) {
      const int j = jc * 2 + t;
      bf16x16 ka0, ka1;
      {
        const float* kp = kbase + (size_t)(j * 16 + lo) * DM_;
        #pragma unroll
        for (int e = 0; e < 8; ++e) {
          ka0[e]     = (__bf16)kp[hi * 8 + e];
          ka0[8 + e] = (__bf16)kp[16 + hi * 8 + e];
          ka1[e]     = (__bf16)kp[32 + hi * 8 + e];
          ka1[8 + e] = (__bf16)kp[48 + hi * 8 + e];
        }
      }
      f32x8 c = {};
      c = wmma_bf16(ka0, bq[0], c);
      c = wmma_bf16(ka1, bq[1], c);
      // element (k = t*16 + r + 8*hi, q = lo); no shuffles needed
      #pragma unroll
      for (int r = 0; r < 8; ++r)
        pw[lo * PST + t * 16 + r + 8 * hi] = __expf(c[r] - mrun) * rcpl;
    }

    // P chunk (16q x 32k, f32 in LDS) -> bf16 A-fragment (same-wave RAW)
    bf16x16 ap;
    #pragma unroll
    for (int e = 0; e < 8; ++e) {
      ap[e]     = (__bf16)pw[lo * PST + 8 * hi + e];
      ap[8 + e] = (__bf16)pw[lo * PST + 16 + 8 * hi + e];
    }

    // coalesced non-temporal attn stores: 16 rows x 32 cols as 128-bit vectors
    #pragma unroll
    for (int it = 0; it < 4; ++it) {
      const int rr = it * 4 + (lane >> 3);
      const int c4 = (lane & 7) * 4;
      const f32x4 pv = *reinterpret_cast<const f32x4*>(&pw[rr * PST + c4]);
      __builtin_nontemporal_store(
          pv, reinterpret_cast<f32x4*>(&attn_base[(size_t)rr * L_ + jc * 32 + c4]));
    }

    // V B-fragments (32 k-rows, four 16-col groups) and P @ V
    #pragma unroll
    for (int og = 0; og < 4; ++og) {
      bf16x16 bv;
      const float* vp = vbase + (size_t)(jc * 32 + hi * 8) * DM_ + og * 16 + lo;
      #pragma unroll
      for (int e = 0; e < 16; ++e) bv[e] = (__bf16)vp[(size_t)((e & 7) + ((e >> 3) << 4)) * DM_];
      oacc[og] = wmma_bf16(ap, bv, oacc[og]);
    }
  }

  // ---- store O stripe into heads workspace [B, L, H*DV] ----
  #pragma unroll
  for (int og = 0; og < 4; ++og) {
    #pragma unroll
    for (int r = 0; r < 8; ++r) {
      const int row = r + 8 * hi;
      heads[((size_t)b * L_ + q0 + row) * DM_ + h * DV_ + og * 16 + lo] =
          oacc[og][r];
    }
  }
}

// ---------------------------------------------------------------------------
// Kernel 2: fc_out = heads @ W_fc + b_fc + residual(q).  One wave per 16x16
// output tile; K = 1024 in 32 bf16-WMMA chunks.
// ---------------------------------------------------------------------------
__global__ void __launch_bounds__(256)
fc_kernel(const float* __restrict__ heads, const float* __restrict__ Wfc,
          const float* __restrict__ bfc, const float* __restrict__ resid,
          float* __restrict__ fcout) {
  const int w    = threadIdx.x >> 5;
  const int lane = threadIdx.x & 31;
  const int lo   = lane & 15;
  const int hi   = lane >> 4;

  const int tile = blockIdx.x * 8 + w;     // 16384 tiles
  const int nt = tile & 63;
  const int mt = tile >> 6;
  const int m0 = mt * 16, n0 = nt * 16;

  f32x8 c = {};
  const float* arow = heads + (size_t)(m0 + lo) * DM_;
  for (int k0 = 0; k0 < DM_; k0 += 32) {
    bf16x16 aa;
    #pragma unroll
    for (int e = 0; e < 8; ++e) {
      aa[e]     = (__bf16)arow[k0 + hi * 8 + e];
      aa[8 + e] = (__bf16)arow[k0 + 16 + hi * 8 + e];
    }
    bf16x16 bb;
    const float* wp = Wfc + (size_t)(k0 + hi * 8) * DM_ + n0 + lo;
    #pragma unroll
    for (int e = 0; e < 16; ++e) bb[e] = (__bf16)wp[(size_t)((e & 7) + ((e >> 3) << 4)) * DM_];
    c = wmma_bf16(aa, bb, c);
  }

  #pragma unroll
  for (int r = 0; r < 8; ++r) {
    const int row = m0 + r + 8 * hi;
    const int col = n0 + lo;
    fcout[(size_t)row * DM_ + col] =
        c[r] + bfc[col] + resid[(size_t)row * DM_ + col];
  }
}

// ---------------------------------------------------------------------------
// Kernel 3: LayerNorm over d_model; one wave per row.
// ---------------------------------------------------------------------------
__global__ void __launch_bounds__(256)
ln_kernel(const float* __restrict__ x, const float* __restrict__ gamma,
          const float* __restrict__ beta, float* __restrict__ out) {
  const int row  = blockIdx.x * 8 + (threadIdx.x >> 5);
  const int lane = threadIdx.x & 31;
  const float* xr = x + (size_t)row * DM_;

  float s = 0.f, s2 = 0.f;
  #pragma unroll
  for (int i = lane; i < DM_; i += 32) {
    const float v = xr[i];
    s += v; s2 += v * v;
  }
  #pragma unroll
  for (int m = 1; m < 32; m <<= 1) { s += __shfl_xor(s, m); s2 += __shfl_xor(s2, m); }
  const float mu  = s * (1.0f / DM_);
  const float var = s2 * (1.0f / DM_) - mu * mu;
  const float rstd = rsqrtf(var + LN_EPS);

  float* orow = out + (size_t)row * DM_;
  #pragma unroll
  for (int i = lane; i < DM_; i += 32)
    orow[i] = (xr[i] - mu) * rstd * gamma[i] + beta[i];
}

// ---------------------------------------------------------------------------
extern "C" void kernel_launch(void* const* d_in, const int* in_sizes, int n_in,
                              void* d_out, int out_size, void* d_ws,
                              size_t ws_size, hipStream_t stream) {
  const float* q     = (const float*)d_in[0];
  const float* k     = (const float*)d_in[1];
  const float* v     = (const float*)d_in[2];
  const float* Wfc   = (const float*)d_in[3];
  const float* bfc   = (const float*)d_in[4];
  const float* gamma = (const float*)d_in[5];
  const float* beta  = (const float*)d_in[6];

  float* out  = (float*)d_out;                       // [B, L, DM]
  float* attn = out + (size_t)B_ * L_ * DM_;         // [B, H, L, L]

  float* heads = (float*)d_ws;                       // [B, L, DM]
  float* fcbuf = heads + (size_t)B_ * L_ * DM_;      // [B, L, DM]

  // 4096 q-stripes, 4 waves/block (9 KB LDS/block)
  attn_kernel<<<1024, 128, 0, stream>>>(q, k, v, attn, heads);
  // 16384 fc tiles, 8 waves/block
  fc_kernel<<<2048, 256, 0, stream>>>(heads, Wfc, bfc, q, fcbuf);
  // 4096 rows, 8 waves/block
  ln_kernel<<<512, 256, 0, stream>>>(fcbuf, gamma, beta, out);
}
